// DualTransformerBlock_68959994904753
// MI455X (gfx1250) — matrix operand order repaired and checked by reading, not compiled, not run on hardware
//
#include <hip/hip_runtime.h>
#include <math.h>

typedef __attribute__((ext_vector_type(16))) _Float16 v16h;
typedef __attribute__((ext_vector_type(8)))  _Float16 v8h;
typedef __attribute__((ext_vector_type(8)))  float    v8f;

#define BB 8
#define CC 256
#define HH 64
#define WW 64
#define NN 4096
#define MM 32768   // BB*NN

// ---------------------------------------------------------------------------
// WMMA fragment helpers (wave32, 16x16x32 f16 -> f32), all-f16 operands.
// A 16x32: lane row = lane&15, khalf = lane>>4; element e -> k = khalf*16 + e (contiguous)
// B 32x16: lane col = lane&15, khalf = lane>>4; element e -> k = khalf*16 + e (contiguous)
// C/D 16x16 f32: elem r -> row = 8*khalf + r, col = lane&15
// ---------------------------------------------------------------------------
__device__ inline v16h load_ah(const _Float16* __restrict__ rowp, int khalf) {
  const v8h lo = *(const v8h*)(rowp + (khalf << 4));
  const v8h hi = *(const v8h*)(rowp + (khalf << 4) + 8);
  v16h r;
#pragma unroll
  for (int e = 0; e < 8; ++e) { r[e] = lo[e]; r[e + 8] = hi[e]; }
  return r;
}
__device__ inline v16h load_bh(const _Float16* __restrict__ p) {
  return *(const v16h*)p;   // 32B-aligned by construction -> 2x global_load_b128
}
__device__ inline v8f wmma_step(v16h a, v16h b, v8f c) {
  return __builtin_amdgcn_wmma_f32_16x16x32_f16(false, a, false, b, (short)0, c, false, false);
}

__device__ inline float gelu_exact(float x) {
  return 0.5f * x * (1.0f + erff(x * 0.70710678118654752f));
}
__device__ inline float sigmoidf_(float x) { return 1.0f / (1.0f + expf(-x)); }

// ---------------------------------------------------------------------------
// GEMM: Y[m,n] = act( A[m,:K] . W[n,:K] + bias[n] ) (+ R[m,n])
// A,W f16; accum f32; optional f32 and/or f16 outputs. Wave tile: 16(M) x 128(N).
// ---------------------------------------------------------------------------
__global__ __launch_bounds__(256) void gemm_wmma(
    const _Float16* __restrict__ A, int lda,
    const _Float16* __restrict__ W,
    const float* __restrict__ bias,
    const float* __restrict__ R, int ldr,
    float* __restrict__ Yf, _Float16* __restrict__ Yh, int ldc,
    int M, int Nout, int K, int act)
{
  const int wave = threadIdx.x >> 5;
  const int lane = threadIdx.x & 31;
  const int tiles_n = Nout >> 7;
  const long tile = (long)blockIdx.x * 8 + wave;
  const long total = (long)(M >> 4) * tiles_n;
  if (tile >= total) return;                 // uniform per wave -> EXEC stays full
  const int tn = (int)(tile % tiles_n);
  const int tm = (int)(tile / tiles_n);
  const int khalf = lane >> 4;
  const int l15 = lane & 15;

  const _Float16* arow = A + (long)((tm << 4) + l15) * lda;
  v8f acc[8] = {};

  for (int k0 = 0; k0 < K; k0 += 32) {
    v16h a = load_ah(arow + k0, khalf);
#pragma unroll
    for (int j = 0; j < 8; ++j) {
      const int col = (tn << 7) + (j << 4) + l15;
      v16h b = load_bh(W + (long)col * K + k0 + (khalf << 4));
      acc[j] = wmma_step(a, b, acc[j]);
    }
  }

  const int mbase = (tm << 4) + (khalf << 3);
#pragma unroll
  for (int j = 0; j < 8; ++j) {
    const int col = (tn << 7) + (j << 4) + l15;
    const float bb = bias ? bias[col] : 0.0f;
#pragma unroll
    for (int r = 0; r < 8; ++r) {
      float v = acc[j][r] + bb;
      if (act == 1) v = fmaxf(v, 0.0f);
      const long ro = mbase + r;
      if (R)  v += R[ro * (long)ldr + col];
      if (Yf) Yf[ro * (long)ldc + col] = v;
      if (Yh) Yh[ro * (long)ldc + col] = (_Float16)v;
    }
  }
}

// ---------------------------------------------------------------------------
// 3x3 SAME conv as 9-tap accumulated WMMA GEMM over token-major [M, Cio] f16.
// W9 pre-packed f16 [9][Cout][Cin].
// ---------------------------------------------------------------------------
__global__ __launch_bounds__(256) void conv3x3_wmma(
    const _Float16* __restrict__ X, const _Float16* __restrict__ W9,
    const float* __restrict__ bias,
    float* __restrict__ Yf, _Float16* __restrict__ Yh,
    int Cio, int act)
{
  const int wave = threadIdx.x >> 5;
  const int lane = threadIdx.x & 31;
  const int tiles_n = Cio >> 7;
  const long tile = (long)blockIdx.x * 8 + wave;
  const long total = (long)(MM >> 4) * tiles_n;
  if (tile >= total) return;
  const int tn = (int)(tile % tiles_n);
  const int tm = (int)(tile / tiles_n);
  const int khalf = lane >> 4;
  const int l15 = lane & 15;

  const int row = (tm << 4) + l15;
  const int bimg = row >> 12;
  const int h = (row >> 6) & 63;
  const int w = row & 63;

  v8f acc[8] = {};

  for (int tap = 0; tap < 9; ++tap) {
    const int dh = tap / 3 - 1, dw = tap % 3 - 1;
    const int hh = h + dh, ww = w + dw;
    const bool valid = (hh >= 0) && (hh < HH) && (ww >= 0) && (ww < WW);
    const _Float16* arow = X + ((long)bimg * NN + hh * WW + ww) * Cio;
    const _Float16* wt = W9 + (long)tap * Cio * Cio;
    for (int k0 = 0; k0 < Cio; k0 += 32) {
      v16h a = v16h{};
      if (valid) a = load_ah(arow + k0, khalf);
#pragma unroll
      for (int j = 0; j < 8; ++j) {
        const int col = (tn << 7) + (j << 4) + l15;
        v16h b = load_bh(wt + (long)col * Cio + k0 + (khalf << 4));
        acc[j] = wmma_step(a, b, acc[j]);
      }
    }
  }

  const int mbase = (tm << 4) + (khalf << 3);
#pragma unroll
  for (int j = 0; j < 8; ++j) {
    const int col = (tn << 7) + (j << 4) + l15;
    const float bb = bias[col];
#pragma unroll
    for (int r = 0; r < 8; ++r) {
      float v = acc[j][r] + bb;
      if (act == 1) v = fmaxf(v, 0.0f);
      if (Yf) Yf[(long)(mbase + r) * Cio + col] = v;
      if (Yh) Yh[(long)(mbase + r) * Cio + col] = (_Float16)v;
    }
  }
}

// f32 -> f16 identity pack (linear weights)
__global__ void pack_h(const float* __restrict__ W, _Float16* __restrict__ Wh, long n) {
  long t = (long)blockIdx.x * 256 + threadIdx.x;
  if (t < n) Wh[t] = (_Float16)W[t];
}

// [O][C][3][3] f32 -> [9][O][C] f16
__global__ void pack_w9h(const float* __restrict__ W, _Float16* __restrict__ W9, int O, int Cc) {
  long t = (long)blockIdx.x * 256 + threadIdx.x;
  long n = (long)O * Cc * 9;
  if (t >= n) return;
  int tap = (int)(t % 9);
  long oc = t / 9;
  int c = (int)(oc % Cc);
  int o = (int)(oc / Cc);
  W9[((long)tap * O + o) * Cc + c] = (_Float16)W[t];
}

// ---------------------------------------------------------------------------
// LayerNorm over C=256 (one wave per token). Optional per-(b,c) pre-scale.
// Writes f32 and/or f16 views.
// ---------------------------------------------------------------------------
__global__ __launch_bounds__(256) void ln_kernel(
    const float* __restrict__ X, const float* __restrict__ g, const float* __restrict__ b,
    const float* __restrict__ scale,
    float* __restrict__ Yf, _Float16* __restrict__ Yh, int Mtot)
{
  const int warp = threadIdx.x >> 5, lane = threadIdx.x & 31;
  const long m = (long)blockIdx.x * 8 + warp;
  if (m >= Mtot) return;
  const int bidx = (int)(m >> 12);
  const float* x = X + m * CC;
  float v[8], s = 0.f, ss = 0.f;
#pragma unroll
  for (int i = 0; i < 8; ++i) {
    const int c = lane + i * 32;
    float t = x[c];
    if (scale) t *= scale[bidx * CC + c];
    v[i] = t; s += t; ss += t * t;
  }
#pragma unroll
  for (int o = 16; o > 0; o >>= 1) { s += __shfl_xor(s, o); ss += __shfl_xor(ss, o); }
  const float mean = s * (1.0f / CC);
  const float var = ss * (1.0f / CC) - mean * mean;
  const float rstd = rsqrtf(var + 1e-5f);
#pragma unroll
  for (int i = 0; i < 8; ++i) {
    const int c = lane + i * 32;
    const float o = (v[i] - mean) * rstd * g[c] + b[c];
    if (Yf) Yf[m * CC + c] = o;
    if (Yh) Yh[m * CC + c] = (_Float16)o;
  }
}

// ---------------------------------------------------------------------------
// Channel attention core, one batch slice. blockIdx.x = c in [0,256).
// ---------------------------------------------------------------------------
__device__ inline float blk_red(float v, float* red, int op) {
#pragma unroll
  for (int o = 16; o > 0; o >>= 1) {
    float t = __shfl_xor(v, o);
    v = op ? fmaxf(v, t) : (v + t);
  }
  __syncthreads();
  if ((threadIdx.x & 31) == 0) red[threadIdx.x >> 5] = v;
  __syncthreads();
  float r = red[0];
#pragma unroll
  for (int i = 1; i < 8; ++i) r = op ? fmaxf(r, red[i]) : (r + red[i]);
  return r;
}

__global__ __launch_bounds__(256) void ca_core(
    const float* __restrict__ QKV, const float* __restrict__ temp8, _Float16* __restrict__ OUT)
{
  __shared__ float sp[NN];
  __shared__ float red[8];
  const int c = blockIdx.x;
  const int head = c >> 5;
  const int tid = threadIdx.x;

  float sqq = 0.f, sqk = 0.f;
  for (int n = tid; n < NN; n += 256) {
    const float q = QKV[(long)n * 768 + c];
    const float k = QKV[(long)n * 768 + 256 + c];
    sp[n] = q * k;
    sqq += q * q; sqk += k * k;
  }
  const float tq = blk_red(sqq, red, 0);
  const float tk = blk_red(sqk, red, 0);
  const float sc = (1.0f / fmaxf(sqrtf(tq), 1e-12f)) *
                   (1.0f / fmaxf(sqrtf(tk), 1e-12f)) * temp8[head];
  __syncthreads();

  float mx = -3.4e38f;
  for (int n = tid; n < NN; n += 256) {
    const float p = sp[n] * sc;
    sp[n] = p;
    mx = fmaxf(mx, p);
  }
  mx = blk_red(mx, red, 1);
  __syncthreads();

  float sum = 0.f;
  for (int n = tid; n < NN; n += 256) {
    const float e = expf(sp[n] - mx);
    sp[n] = e;
    sum += e;
  }
  sum = blk_red(sum, red, 0);
  const float inv = 1.0f / sum;
  __syncthreads();

  for (int n = tid; n < NN; n += 256)
    OUT[(long)n * CC + c] = (_Float16)(sp[n] * inv * QKV[(long)n * 768 + 512 + c]);
}

// softmax over channels (axis C), one wave per token, in place (f32)
__global__ __launch_bounds__(256) void softmax_c(float* __restrict__ X, int Mtot) {
  const int warp = threadIdx.x >> 5, lane = threadIdx.x & 31;
  const long m = (long)blockIdx.x * 8 + warp;
  if (m >= Mtot) return;
  float* x = X + m * CC;
  float v[8], mx = -3.4e38f;
#pragma unroll
  for (int i = 0; i < 8; ++i) { v[i] = x[lane + i * 32]; mx = fmaxf(mx, v[i]); }
#pragma unroll
  for (int o = 16; o > 0; o >>= 1) mx = fmaxf(mx, __shfl_xor(mx, o));
  float s = 0.f;
#pragma unroll
  for (int i = 0; i < 8; ++i) { v[i] = expf(v[i] - mx); s += v[i]; }
#pragma unroll
  for (int o = 16; o > 0; o >>= 1) s += __shfl_xor(s, o);
  const float inv = 1.0f / s;
#pragma unroll
  for (int i = 0; i < 8; ++i) x[lane + i * 32] = v[i] * inv;
}

// softmax over H axis for fixed (b, w, c), in place (f32)
__global__ __launch_bounds__(256) void softmax_h(float* __restrict__ X) {
  const long t = (long)blockIdx.x * 256 + threadIdx.x;   // over 8*64*256
  const int c = (int)(t & 255);
  const int w = (int)((t >> 8) & 63);
  const int b = (int)(t >> 14);
  float* p = X + ((long)b * NN + w) * CC + c;
  const long stride = (long)WW * CC;
  float mx = -3.4e38f;
  for (int h = 0; h < HH; ++h) mx = fmaxf(mx, p[h * stride]);
  float s = 0.f;
  for (int h = 0; h < HH; ++h) s += expf(p[h * stride] - mx);
  const float inv = 1.0f / s;
  for (int h = 0; h < HH; ++h) p[h * stride] = expf(p[h * stride] - mx) * inv;
}

// att = k * v * q * temp[c]  -> f16 (GEMM operand)
__global__ void att_mul(const float* __restrict__ K_, const float* __restrict__ V_,
                        const float* __restrict__ Q_, const float* __restrict__ T_,
                        _Float16* __restrict__ Y, long total) {
  const long i = (long)blockIdx.x * 256 + threadIdx.x;
  if (i >= total) return;
  Y[i] = (_Float16)(K_[i] * V_[i] * Q_[i] * T_[i & 255]);
}

// SE: partial sums over 256-token chunks. block = b*16 + chunk
__global__ __launch_bounds__(256) void se_partial(const float* __restrict__ X, float* __restrict__ part) {
  const int b = blockIdx.x >> 4, ch = blockIdx.x & 15, c = threadIdx.x;
  float s = 0.f;
  const long base = ((long)b * NN + ch * 256) * CC + c;
  for (int n = 0; n < 256; ++n) s += X[base + (long)n * CC];
  part[(long)blockIdx.x * CC + c] = s;
}

// SE MLP: scale[b,c] = 1 + sigmoid(gelu(mean@red^T+rb)@exp^T+eb)
__global__ __launch_bounds__(256) void se_mlp(
    const float* __restrict__ part, const float* __restrict__ red_w, const float* __restrict__ red_b,
    const float* __restrict__ exp_w, const float* __restrict__ exp_b, float* __restrict__ scale)
{
  __shared__ float mean[CC];
  __shared__ float s1[64];
  const int b = blockIdx.x, c = threadIdx.x;
  float s = 0.f;
  for (int ch = 0; ch < 16; ++ch) s += part[((long)b * 16 + ch) * CC + c];
  mean[c] = s * (1.0f / NN);
  __syncthreads();
  if (c < 64) {
    float a = red_b[c];
    for (int j = 0; j < CC; ++j) a += mean[j] * red_w[c * CC + j];
    s1[c] = gelu_exact(a);
  }
  __syncthreads();
  float a2 = exp_b[c];
  for (int j = 0; j < 64; ++j) a2 += s1[j] * exp_w[c * 64 + j];
  scale[(long)b * CC + c] = 1.0f + sigmoidf_(a2);
}

// depthwise 3x3 + gelu on one batch image, token-major [4096, hid], f16 in/out
__global__ __launch_bounds__(256) void dwconv_gelu(
    const _Float16* __restrict__ X, const float* __restrict__ Wd, const float* __restrict__ Bd,
    _Float16* __restrict__ Y, int hid)
{
  const long t = (long)blockIdx.x * 256 + threadIdx.x;
  const long total = (long)NN * hid;
  if (t >= total) return;
  const int c = (int)(t % hid);
  const int n = (int)(t / hid);
  const int h = n >> 6, w = n & 63;
  float s = Bd[c];
#pragma unroll
  for (int tap = 0; tap < 9; ++tap) {
    const int hh = h + tap / 3 - 1, ww = w + tap % 3 - 1;
    if (hh >= 0 && hh < HH && ww >= 0 && ww < WW)
      s += (float)X[(long)(hh * WW + ww) * hid + c] * Wd[c * 9 + tap];
  }
  Y[t] = (_Float16)gelu_exact(s);
}

// bn(eval) + relu: y = relu(x * g/sqrt(1+eps) + b), f32 in -> f16 out (GEMM operand)
__global__ void bnrelu(const float* __restrict__ X, const float* __restrict__ g,
                       const float* __restrict__ b, _Float16* __restrict__ Y, long total, int Cch) {
  const long i = (long)blockIdx.x * 256 + threadIdx.x;
  if (i >= total) return;
  const int c = (int)(i % Cch);
  const float inv = rsqrtf(1.0f + 1e-5f);
  Y[i] = (_Float16)fmaxf(X[i] * inv * g[c] + b[c], 0.0f);
}

// ---------------------------------------------------------------------------
extern "C" void kernel_launch(void* const* d_in, const int* in_sizes, int n_in,
                              void* d_out, int out_size, void* d_ws, size_t ws_size,
                              hipStream_t stream) {
  (void)in_sizes; (void)n_in; (void)out_size; (void)ws_size;
  auto F = [&](int i) { return (const float*)d_in[i]; };

  const float* x      = F(0);
  const float *g1=F(1), *b1=F(2), *g2=F(3), *b2=F(4), *g4=F(5), *b4=F(6),
              *g5=F(7), *b5=F(8), *g6=F(9), *b6=F(10);
  const float *qkv_w=F(11), *ca_temp=F(12), *proj_w=F(13), *proj_b=F(14);
  const float *k_w=F(15), *k_b=F(16), *q_w=F(17), *q_b=F(18),
              *v_w=F(19), *v_b=F(20), *r_w=F(21), *r_b=F(22), *ea_temp=F(23);
  const float *c3_w=F(24), *c3_b=F(25), *c1_w=F(26), *c1_b=F(27);
  const float *red_w=F(28), *red_b=F(29), *exp_w=F(30), *exp_b=F(31);
  const float *bn1_g=F(32), *bn1_b=F(33), *rc1_w=F(34), *rc1_b=F(35),
              *bn2_g=F(36), *bn2_b=F(37), *rc2_w=F(38), *rc2_b=F(39),
              *bn3_g=F(40), *bn3_b=F(41), *rc3_w=F(42), *rc3_b=F(43);
  const float *m1_fc1w=F(44), *m1_fc1b=F(45), *m1_dww=F(46), *m1_dwb=F(47),
              *m1_fc2w=F(48), *m1_fc2b=F(49);
  const float *m2_fc1w=F(50), *m2_fc1b=F(51), *m2_dww=F(52), *m2_dwb=F(53),
              *m2_fc2w=F(54), *m2_fc2b=F(55);
  float* out = (float*)d_out;

  // ---- workspace layout ----
  const long SZ = (long)MM * CC;                  // 8,388,608
  float* T0f = (float*)d_ws;
  float* T1f = T0f + SZ;
  float* T2f = T1f + SZ;
  float* T3f = T2f + SZ;
  float* QKVb = T3f + SZ;                         // [4096,768] f32
  float* PART = QKVb + (long)NN * 768;            // [8*16*256]
  float* SCALE = PART + 8 * 16 * 256;             // [8*256]
  _Float16* T0h = (_Float16*)(SCALE + 8 * 256 + 256);  // pad; base 64B aligned
  _Float16* T1h = T0h + SZ;
  _Float16* T3h = T1h + SZ;
  _Float16* MERGEh = T3h + SZ;                    // [MM,512] f16
  _Float16* FAh = MERGEh + (long)MM * 512;        // [4096,1024]
  _Float16* FBh = FAh + (long)NN * 1024;
  _Float16* WH = FBh + (long)NN * 1024;           // f16 weight arena
  _Float16* qkv_wh = WH;                          long o = 768 * 256;
  _Float16* proj_wh = WH + o;                     o += 256 * 256;
  _Float16* k_wh = WH + o;                        o += 256 * 256;
  _Float16* q_wh = WH + o;                        o += 256 * 256;
  _Float16* v_wh = WH + o;                        o += 256 * 256;
  _Float16* r_wh = WH + o;                        o += 256 * 256;
  _Float16* c1_wh = WH + o;                       o += 256 * 512;
  _Float16* rc1_wh = WH + o;                      o += 128 * 256;
  _Float16* rc3_wh = WH + o;                      o += 256 * 128;
  _Float16* m1_fc1wh = WH + o;                    o += 1024 * 256;
  _Float16* m1_fc2wh = WH + o;                    o += 256 * 1024;
  _Float16* m2_fc1wh = WH + o;                    o += 1024 * 256;
  _Float16* m2_fc2wh = WH + o;                    o += 256 * 1024;
  _Float16* W9Ah = WH + o;                        o += 9 * 256 * 256;
  _Float16* W9Bh = WH + o;

  auto gemm = [&](const _Float16* A, int lda, const _Float16* W, const float* bias,
                  const float* R, int ldr, float* Yf, _Float16* Yh, int ldc,
                  int M, int N, int K, int act) {
    const int tiles = (M >> 4) * (N >> 7);
    const int blocks = (tiles + 7) / 8;
    gemm_wmma<<<blocks, 256, 0, stream>>>(A, lda, W, bias, R, ldr, Yf, Yh, ldc, M, N, K, act);
  };
  auto packh = [&](const float* W, _Float16* Wh, long n) {
    pack_h<<<(int)((n + 255) / 256), 256, 0, stream>>>(W, Wh, n);
  };

  // ---- pack all GEMM/conv weights to f16 (once per launch, tiny) ----
  packh(qkv_w, qkv_wh, 768 * 256);
  packh(proj_w, proj_wh, 256 * 256);
  packh(k_w, k_wh, 256 * 256);  packh(q_w, q_wh, 256 * 256);
  packh(v_w, v_wh, 256 * 256);  packh(r_w, r_wh, 256 * 256);
  packh(c1_w, c1_wh, 256 * 512);
  packh(rc1_w, rc1_wh, 128 * 256);
  packh(rc3_w, rc3_wh, 256 * 128);
  packh(m1_fc1w, m1_fc1wh, 1024 * 256);  packh(m1_fc2w, m1_fc2wh, 256 * 1024);
  packh(m2_fc1w, m2_fc1wh, 1024 * 256);  packh(m2_fc2w, m2_fc2wh, 256 * 1024);
  pack_w9h<<<(256 * 256 * 9 + 255) / 256, 256, 0, stream>>>(c3_w, W9Ah, 256, 256);
  pack_w9h<<<(128 * 128 * 9 + 255) / 256, 256, 0, stream>>>(rc2_w, W9Bh, 128, 128);

  // ---------- branch A: channel attention + SE + MixFFN2 ----------
  ln_kernel<<<MM / 8, 256, 0, stream>>>(x, g5, b5, nullptr, T0f, T0h, MM);      // seg
  for (int b = 0; b < BB; ++b) {
    gemm(T0h + (long)b * NN * CC, CC, qkv_wh, nullptr, nullptr, 0,
         QKVb, nullptr, 768, NN, 768, CC, 0);                                   // qkv (batch b)
    ca_core<<<CC, 256, 0, stream>>>(QKVb, ca_temp, T1h + (long)b * NN * CC);
  }
  gemm(T1h, CC, proj_wh, proj_b, T0f, CC, T2f, nullptr, CC, MM, CC, CC, 0);     // add3 -> T2f
  ln_kernel<<<MM / 8, 256, 0, stream>>>(T2f, g4, b4, nullptr, T0f, nullptr, MM); // n4 -> T0f
  se_partial<<<BB * 16, 256, 0, stream>>>(T0f, PART);
  se_mlp<<<BB, 256, 0, stream>>>(PART, red_w, red_b, exp_w, exp_b, SCALE);
  ln_kernel<<<MM / 8, 256, 0, stream>>>(T0f, g6, b6, SCALE, T1f, T1h, MM);      // se_out
  for (int b = 0; b < BB; ++b) {
    gemm(T1h + (long)b * NN * CC, CC, m2_fc1wh, m2_fc1b, nullptr, 0,
         nullptr, FAh, 1024, NN, 1024, CC, 0);
    dwconv_gelu<<<(NN * 1024 + 255) / 256, 256, 0, stream>>>(FAh, m2_dww, m2_dwb, FBh, 1024);
    gemm(FBh, 1024, m2_fc2wh, m2_fc2b, T1f + (long)b * NN * CC, CC,
         nullptr, MERGEh + (long)b * NN * 512 + 256, 512, NN, CC, 1024, 0);     // mlp2_out
  }

  // ---------- branch B: efficient attention + MixFFN1 ----------
  ln_kernel<<<MM / 8, 256, 0, stream>>>(x, g1, b1, nullptr, nullptr, T0h, MM);  // n1
  conv3x3_wmma<<<(MM / 16 * 2 + 7) / 8, 256, 0, stream>>>(T0h, W9Ah, c3_b,
      nullptr, T1h, CC, 1);                                                     // relu conv
  gemm(T1h, CC, k_wh, k_b, nullptr, 0, T2f, nullptr, CC, MM, CC, CC, 0);
  softmax_h<<<(BB * WW * CC) / 256, 256, 0, stream>>>(T2f);                     // k
  gemm(T1h, CC, q_wh, q_b, nullptr, 0, T0f, nullptr, CC, MM, CC, CC, 0);
  softmax_c<<<MM / 8, 256, 0, stream>>>(T0f, MM);                               // q
  gemm(T1h, CC, v_wh, v_b, nullptr, 0, T3f, nullptr, CC, MM, CC, CC, 0);        // v
  att_mul<<<(int)(SZ / 256), 256, 0, stream>>>(T2f, T3f, T0f, ea_temp, T1h, SZ);
  gemm(T1h, CC, r_wh, r_b, x, CC, T2f, nullptr, CC, MM, CC, CC, 0);             // add1 -> T2f
  ln_kernel<<<MM / 8, 256, 0, stream>>>(T2f, g2, b2, nullptr, nullptr, T0h, MM); // ln2
  for (int b = 0; b < BB; ++b) {
    gemm(T0h + (long)b * NN * CC, CC, m1_fc1wh, m1_fc1b, nullptr, 0,
         nullptr, FAh, 1024, NN, 1024, CC, 0);
    dwconv_gelu<<<(NN * 1024 + 255) / 256, 256, 0, stream>>>(FAh, m1_dww, m1_dwb, FBh, 1024);
    gemm(FBh, 1024, m1_fc2wh, m1_fc2b, T2f + (long)b * NN * CC, CC,
         nullptr, MERGEh + (long)b * NN * 512, 512, NN, CC, 1024, 0);           // add2
  }

  // ---------- merge + residual branch ----------
  gemm(MERGEh, 512, c1_wh, c1_b, nullptr, 0, T0f, nullptr, CC, MM, CC, 512, 1); // merge_out
  bnrelu<<<(int)(SZ / 256), 256, 0, stream>>>(T0f, bn1_g, bn1_b, T1h, SZ, CC);
  gemm(T1h, CC, rc1_wh, rc1_b, nullptr, 0, T2f, nullptr, 128, MM, 128, CC, 0);
  bnrelu<<<(int)((long)MM * 128 / 256), 256, 0, stream>>>(T2f, bn2_g, bn2_b, T3h, (long)MM * 128, 128);
  conv3x3_wmma<<<(MM / 16 * 1 + 7) / 8, 256, 0, stream>>>(T3h, W9Bh, rc2_b,
      T3f, nullptr, 128, 0);
  bnrelu<<<(int)((long)MM * 128 / 256), 256, 0, stream>>>(T3f, bn3_g, bn3_b, T1h, (long)MM * 128, 128);
  gemm(T1h, 128, rc3_wh, rc3_b, T0f, CC, out, nullptr, CC, MM, CC, 128, 0);     // out = merge + res
}
